// MLP_18631568130181
// MI455X (gfx1250) — matrix operand order repaired and checked by reading, not compiled, not run on hardware
//
#include <hip/hip_runtime.h>

// ---- CDNA5 WMMA vector types -------------------------------------------------
typedef __attribute__((ext_vector_type(16))) _Float16 v16h;
typedef __attribute__((ext_vector_type(8)))  _Float16 v8h;
typedef __attribute__((ext_vector_type(8)))  float    v8f;

#define HDIM        512
#define EDIM        8
#define OUTD        2
#define ROWS_PER_WG 64
#define H_STRIDE    520   // LDS row stride in halves (512 + 8 pad -> bank spread)

// -----------------------------------------------------------------------------
// Kernel 1: w2 (512x512 f32, row-major [k][n]) -> w2T (f16, [n][k]).
// Lets each B-matrix WMMA fragment be a contiguous 32B per-lane load:
// lane L reads column n = nb + (L&15), K = ks + (L>=16 ? 16 : 0) .. +15.
// -----------------------------------------------------------------------------
__global__ __launch_bounds__(256) void w2_to_f16_T(const float* __restrict__ w2,
                                                   _Float16* __restrict__ w2t) {
    int i = blockIdx.x * blockDim.x + threadIdx.x;   // 0 .. 512*512-1
    int k = i >> 9;
    int n = i & (HDIM - 1);
    w2t[n * HDIM + k] = (_Float16)w2[i];
}

// -----------------------------------------------------------------------------
// Kernel 2: fully fused MLP for a 64-row tile.
// Each wave computes the full 64x64 output tile of hidden2: 16 f32 accumulators
// (4 M-tiles x 4 N-tiles), 16 WMMAs per K-step on 4 A-frags + 4 B-frags.
// -----------------------------------------------------------------------------
__global__ __launch_bounds__(256) void fused_mlp(
    const float* __restrict__ x,       // [N,10]
    const float* __restrict__ w1,      // [512]
    const float* __restrict__ b1,      // [512]
    const float* __restrict__ meta_w,  // [8,512]
    const float* __restrict__ meta_b,  // [8,512]
    const _Float16* __restrict__ w2t,  // [512,512] f16 transposed ([n][k])
    const float* __restrict__ b2,      // [512]
    const float* __restrict__ w3,      // [512,2]
    const float* __restrict__ b3,      // [2]
    float* __restrict__ out)           // [N,2]
{
    extern __shared__ char smem[];
    _Float16* h1 = (_Float16*)smem;                                   // [64][520] f16
    _Float16* h2 = (_Float16*)(smem + ROWS_PER_WG * H_STRIDE * 2);    // [64][520] f16
    float*    xr = (float*)   (smem + 2 * ROWS_PER_WG * H_STRIDE * 2);// [64][10]  f32

    const int tid = threadIdx.x;
    const int r0  = blockIdx.x * ROWS_PER_WG;

    // ---- stage x rows (one_hot[8], delta, phi per row) into LDS -------------
    for (int i = tid; i < ROWS_PER_WG * 10; i += 256)
        xr[i] = x[r0 * 10 + i];
    __syncthreads();

    // ---- Phase A: h1 = relu(delta*w1 + b1 + phi*(oh.meta_w) + oh.meta_b) ----
    {
        float w1c[2], b1c[2], mwc[2][EDIM], mbc[2][EDIM];
        #pragma unroll
        for (int j = 0; j < 2; ++j) {
            const int c = tid + j * 256;
            w1c[j] = w1[c];
            b1c[j] = b1[c];
            #pragma unroll
            for (int e = 0; e < EDIM; ++e) {
                mwc[j][e] = meta_w[e * HDIM + c];
                mbc[j][e] = meta_b[e * HDIM + c];
            }
        }
        for (int r = 0; r < ROWS_PER_WG; ++r) {
            const float* row = &xr[r * 10];       // broadcast reads from LDS
            const float delta = row[8];
            const float phi   = row[9];
            #pragma unroll
            for (int j = 0; j < 2; ++j) {
                const int c = tid + j * 256;
                float s = 0.f, t = 0.f;
                #pragma unroll
                for (int e = 0; e < EDIM; ++e) {
                    s = fmaf(row[e], mwc[j][e], s);
                    t = fmaf(row[e], mbc[j][e], t);
                }
                float v = fmaf(delta, w1c[j], b1c[j]) + fmaf(phi, s, t);
                v = v > 0.f ? v : 0.f;
                h1[r * H_STRIDE + c] = (_Float16)v;
            }
        }
    }
    __syncthreads();

    // ---- Phase B: h2 = relu(h1 @ w2 + b2) on the WMMA engine ----------------
    {
        const int wave  = tid >> 5;
        const int lane  = tid & 31;
        const int lrow  = lane & 15;      // A: row within tile / B,C,D: column
        const int hi    = lane >> 4;      // lane half
        const int akh   = hi * 16;        // A K-offset base (contiguous K, matches B)
        const int bkh   = hi * 16;        // B K-offset base
        const int nbase = wave * 64;      // this wave's 64 output columns

        v8f acc[4][4] = {};               // [nt][mt] -> 64x64 f32 tile per wave

        for (int ks = 0; ks < HDIM; ks += 32) {
            // 4 A fragments (one per M-tile) from LDS, loaded once per K-step
            union { v16h v; v8h h[2]; } af[4];
            #pragma unroll
            for (int mt = 0; mt < 4; ++mt) {
                const int arow = mt * 16 + lrow;
                af[mt].h[0] = *(const v8h*)(&h1[arow * H_STRIDE + ks + akh]);
                af[mt].h[1] = *(const v8h*)(&h1[arow * H_STRIDE + ks + akh + 8]);
            }
            // 4 B fragments (one per N-tile): contiguous 32B/lane from f16 w2T
            v16h bf[4];
            #pragma unroll
            for (int nt = 0; nt < 4; ++nt)
                bf[nt] = *(const v16h*)(&w2t[(nbase + nt * 16 + lrow) * HDIM + ks + bkh]);
            // 16 back-to-back WMMAs, no memory ops interleaved
            #pragma unroll
            for (int nt = 0; nt < 4; ++nt)
                #pragma unroll
                for (int mt = 0; mt < 4; ++mt)
                    acc[nt][mt] = __builtin_amdgcn_wmma_f32_16x16x32_f16(
                        /*neg_a=*/false, af[mt].v, /*neg_b=*/false, bf[nt],
                        /*c_mod=*/(short)0, acc[nt][mt],
                        /*reuse_a=*/false, /*reuse_b=*/false);
        }

        // epilogue: +b2, relu, store f16 h2 using the C/D VGPR layout
        #pragma unroll
        for (int nt = 0; nt < 4; ++nt) {
            const int ncol = nbase + nt * 16 + lrow;
            const float bias = b2[ncol];
            #pragma unroll
            for (int mt = 0; mt < 4; ++mt) {
                #pragma unroll
                for (int v = 0; v < 8; ++v) {
                    const int rrow = mt * 16 + (hi ? v + 8 : v);
                    float val = acc[nt][mt][v] + bias;
                    val = val > 0.f ? val : 0.f;
                    h2[rrow * H_STRIDE + ncol] = (_Float16)val;
                }
            }
        }
    }
    __syncthreads();

    // ---- Phase C: out = h2 @ w3 + b3 (512x2: plain VALU dot products) -------
    if (tid < ROWS_PER_WG * OUTD) {
        const int r = tid >> 1;
        const int o = tid & 1;
        float s = b3[o];
        for (int c = 0; c < HDIM; ++c)
            s = fmaf((float)h2[r * H_STRIDE + c], w3[c * OUTD + o], s);
        out[(r0 + r) * OUTD + o] = s;
    }
}

// -----------------------------------------------------------------------------
extern "C" void kernel_launch(void* const* d_in, const int* in_sizes, int n_in,
                              void* d_out, int out_size, void* d_ws, size_t ws_size,
                              hipStream_t stream) {
    const float* x      = (const float*)d_in[0];
    const float* w1     = (const float*)d_in[1];
    const float* b1     = (const float*)d_in[2];
    const float* meta_w = (const float*)d_in[3];
    const float* meta_b = (const float*)d_in[4];
    const float* w2     = (const float*)d_in[5];
    const float* b2     = (const float*)d_in[6];
    const float* w3     = (const float*)d_in[7];
    const float* b3     = (const float*)d_in[8];
    float* out = (float*)d_out;

    _Float16* w2t = (_Float16*)d_ws;   // 512*512*2 = 512 KB scratch

    const int N = in_sizes[0] / 10;    // 65536

    // 1) convert + transpose w2 to f16 (stays resident in the 192MB L2)
    w2_to_f16_T<<<(HDIM * HDIM) / 256, 256, 0, stream>>>(w2, w2t);

    // 2) fused MLP, 64 rows per workgroup, 8 waves each
    const size_t lds_bytes = 2 * (size_t)ROWS_PER_WG * H_STRIDE * 2  // h1 + h2 (f16)
                           + (size_t)ROWS_PER_WG * 10 * 4;           // staged x rows
    fused_mlp<<<N / ROWS_PER_WG, 256, lds_bytes, stream>>>(
        x, w1, b1, meta_w, meta_b, w2t, b2, w3, b3, out);
}
